// Linear4bit_12403865551685
// MI455X (gfx1250) — compile-verified, matrix operand order checked by reading, not run on hardware
//
#include <hip/hip_runtime.h>
#include <hip/hip_bf16.h>

// Problem constants
#define MM   8192
#define KK   4096
#define NN   12288
#define KT64 64            // K / 64 = number of K-tiles
#define TILE_BYTES 1024    // one 16x64 int8 tile = 32 lanes * 32 bytes

typedef int v8i __attribute__((ext_vector_type(8)));

#if defined(__has_builtin)
#if __has_builtin(__builtin_amdgcn_sched_barrier)
#define SCHED_FENCE() __builtin_amdgcn_sched_barrier(0)
#endif
#endif
#ifndef SCHED_FENCE
#define SCHED_FENCE()
#endif

// ---------------------------------------------------------------------------
// Stage 1a: convert x_q (int32 per int4 value, row-major [M,K]) into the
// fragment-swizzled int8 A layout: tile(mt,kt) -> 32 lanes x 32 contiguous
// bytes, where lane l (row = l&15, half = l>>4) byte-chunk i (0..3) holds
// K = kt*64 + i*16 + half*8 .. +7.   One thread = one 8-byte chunk.
// ---------------------------------------------------------------------------
__global__ __launch_bounds__(256) void pack_x_kernel(const int* __restrict__ xq,
                                                     char* __restrict__ xp) {
    size_t t = (size_t)blockIdx.x * 256 + threadIdx.x;   // [0, M*K/8)
    int i    = (int)(t & 3);
    int lane = (int)((t >> 2) & 31);
    size_t tile = t >> 7;
    int kt = (int)(tile & (KT64 - 1));
    int mt = (int)(tile >> 6);
    int row   = mt * 16 + (lane & 15);
    int kbase = kt * 64 + i * 16 + (lane >> 4) * 8;
    const int* src = xq + (size_t)row * KK + kbase;
    int4 lo = *(const int4*)src;
    int4 hi = *(const int4*)(src + 4);
    unsigned d0 = (lo.x & 255) | ((lo.y & 255) << 8) | ((lo.z & 255) << 16) | ((unsigned)(lo.w & 255) << 24);
    unsigned d1 = (hi.x & 255) | ((hi.y & 255) << 8) | ((hi.z & 255) << 16) | ((unsigned)(hi.w & 255) << 24);
    ((uint2*)xp)[t] = make_uint2(d0, d1);
}

// ---------------------------------------------------------------------------
// Stage 1b: unpack int4 weight nibbles (low nibble = even K) into the same
// fragment-swizzled int8 B layout over [N,K].  One thread = 8 K-values =
// 4 packed bytes (each stored as one int32 in [0,256)).
// ---------------------------------------------------------------------------
__device__ __forceinline__ unsigned unpack2(int b) {
    int lo = ((b & 0xF) ^ 8) - 8;          // sign-extend low nibble  (even K)
    int hi = (((b >> 4) & 0xF) ^ 8) - 8;   // sign-extend high nibble (odd K)
    return (unsigned)(lo & 0xFF) | ((unsigned)(hi & 0xFF) << 8);
}

__global__ __launch_bounds__(256) void pack_w_kernel(const int* __restrict__ w,
                                                     char* __restrict__ wp) {
    size_t t = (size_t)blockIdx.x * 256 + threadIdx.x;   // [0, N*K/8)
    int i    = (int)(t & 3);
    int lane = (int)((t >> 2) & 31);
    size_t tile = t >> 7;
    int kt = (int)(tile & (KT64 - 1));
    int nt = (int)(tile >> 6);
    int row   = nt * 16 + (lane & 15);
    int kbase = kt * 64 + i * 16 + (lane >> 4) * 8;      // even
    const int* src = w + (size_t)row * (KK / 2) + (kbase >> 1);
    int4 b = *(const int4*)src;
    unsigned d0 = unpack2(b.x) | (unpack2(b.y) << 16);
    unsigned d1 = unpack2(b.z) | (unpack2(b.w) << 16);
    ((uint2*)wp)[t] = make_uint2(d0, d1);
}

// ---------------------------------------------------------------------------
// Stage 2: WMMA-IU8 GEMM.  Block = 256 threads = 8 waves, block tile 128x256,
// wave tile 64x64 = 4x4 v_wmma_i32_16x16x64_iu8 tiles.  All 16 b128 fragment
// loads for a K-step are issued ahead of the 16 WMMAs (sched fence) so waits
// cascade down.  Prefetch uses the SAME base expression as the loads (+8
// tiles, unclamped but provably in-bounds) so it folds to an immediate-offset
// global_prefetch_b8 with no VALU address math -> no WMMA->VALU hazard NOPs.
// ---------------------------------------------------------------------------
__global__ __launch_bounds__(256, 1) void gemm_i4_wmma_kernel(
    const char*  __restrict__ Ap,   // swizzled x8
    const char*  __restrict__ Bp,   // swizzled w8
    const float* __restrict__ sx,   // [M] per-row activation scales
    const float* __restrict__ sw,   // [N] per-output weight scales
    const float* __restrict__ bias, // [N]
    float*       __restrict__ out)  // [M,N] row-major
{
    const int lane = threadIdx.x & 31;
    const int wave = threadIdx.x >> 5;
    const int wrow = wave & 1;      // 2 waves along M
    const int wcol = wave >> 1;     // 4 waves along N
    const int mt0 = blockIdx.x * 8  + wrow * 4;   // 16-row tile index in M
    const int nt0 = blockIdx.y * 16 + wcol * 4;   // 16-col tile index in N

    const char* Abase = Ap + (size_t)mt0 * KT64 * TILE_BYTES;
    const char* Bbase = Bp + (size_t)nt0 * KT64 * TILE_BYTES;
    const int loff = lane * 32;

    v8i acc[4][4] = {};

    for (int kt = 0; kt < KT64; ++kt) {
        const char* arow = Abase + (size_t)kt * TILE_BYTES + loff;
        const char* brow = Bbase + (size_t)kt * TILE_BYTES + loff;

        // Prefetch +8 K-steps: same base as the loads, pure immediate offset.
        // Overshoot past kt=63 lands in the next tile-row of the same packed
        // array (always in-bounds); prefetch is speculative regardless.
        __builtin_prefetch(arow + 8 * TILE_BYTES, 0, 3);
        __builtin_prefetch(brow + 8 * TILE_BYTES, 0, 3);

        v8i af[4], bf[4];
        // Issue order: af0, bf0..3, af1..3  ->  WMMA order (a-outer, b-inner)
        // consumes with descending s_wait_loadcnt cascade.
        af[0] = *(const v8i*)(arow);
#pragma unroll
        for (int b = 0; b < 4; ++b)
            bf[b] = *(const v8i*)(brow + (size_t)b * KT64 * TILE_BYTES);
#pragma unroll
        for (int a = 1; a < 4; ++a)
            af[a] = *(const v8i*)(arow + (size_t)a * KT64 * TILE_BYTES);

        SCHED_FENCE();   // keep all 16 b128 loads ahead of the WMMA stream

#pragma unroll
        for (int a = 0; a < 4; ++a)
#pragma unroll
            for (int b = 0; b < 4; ++b)
                acc[a][b] = __builtin_amdgcn_wmma_i32_16x16x64_iu8(
                    /*sgn_a=*/true, af[a], /*sgn_b=*/true, bf[b],
                    acc[a][b], /*reuse_a=*/false, /*reuse_b=*/false);
    }

    // Epilogue: C layout -> lane l holds column (l&15); VGPR r holds row
    // r + 8*(l>>4).  Dequant + bias, write f32.
    const int nl   = lane & 15;
    const int half = lane >> 4;
    float wsv[4], biv[4];
#pragma unroll
    for (int b = 0; b < 4; ++b) {
        int col = nt0 * 16 + b * 16 + nl;
        wsv[b] = sw[col];
        biv[b] = bias[col];
    }
#pragma unroll
    for (int a = 0; a < 4; ++a) {
        int rowbase = mt0 * 16 + a * 16 + half * 8;
        float sxv[8];
#pragma unroll
        for (int r = 0; r < 8; ++r) sxv[r] = sx[rowbase + r];
#pragma unroll
        for (int b = 0; b < 4; ++b) {
            int col = nt0 * 16 + b * 16 + nl;
            float* op = out + (size_t)rowbase * NN + col;
#pragma unroll
            for (int r = 0; r < 8; ++r)
                op[(size_t)r * NN] = (float)acc[a][b][r] * sxv[r] * wsv[b] + biv[b];
        }
    }
}

// ---------------------------------------------------------------------------
extern "C" void kernel_launch(void* const* d_in, const int* in_sizes, int n_in,
                              void* d_out, int out_size, void* d_ws, size_t ws_size,
                              hipStream_t stream) {
    const int*   xq   = (const int*)d_in[0];
    const float* sx   = (const float*)d_in[1];
    const int*   w    = (const int*)d_in[2];
    const float* sw   = (const float*)d_in[3];
    const float* bias = (const float*)d_in[4];
    float*       out  = (float*)d_out;

    char* xp = (char*)d_ws;                       // 8192*4096  = 32 MiB
    char* wp = xp + (size_t)MM * KK;              // 12288*4096 = 48 MiB

    // M*K/8 chunks / 256 threads
    pack_x_kernel<<<(MM * (size_t)KK / 8) / 256, 256, 0, stream>>>(xq, xp);
    // N*K/8 chunks / 256 threads
    pack_w_kernel<<<(NN * (size_t)KK / 8) / 256, 256, 0, stream>>>(w, wp);

    dim3 grid(MM / 128, NN / 256);                // 64 x 48 blocks
    gemm_i4_wmma_kernel<<<grid, 256, 0, stream>>>(xp, wp, sx, sw, bias, out);
}
